// MoEDeepSeekV3_22153441312859
// MI455X (gfx1250) — compile-verified, matrix operand order checked by reading, not run on hardware
//
#include <hip/hip_runtime.h>
#include <stdint.h>

// MoE DeepSeekV3: T=2048 D=1024 E=16 I=512, groups=4 topk_groups=2 topk=4.
// bf16 WMMA (v_wmma_f32_16x16x32_bf16) for all GEMMs, dense over experts
// (weights are L2-resident: 192MB L2 > 107MB weights), fused silu/gate.
// Staging uses CDNA5 async-to-LDS (global_load_async_to_lds_b128 + ASYNCcnt)
// with double-buffered LDS; each wave register-blocks 2 M-tiles x 1 N-tile.
// Workspace budget ~93MB.

#define TOK 2048
#define DIM 1024
#define NEXP 16
#define IDIM 512
#define NGRP 4
#define NSLOT (NEXP + 1)   // slot 16 = shared expert (gate weight 1.0)

#define MT 64     // token tile (M)
#define IT 64     // intermediate tile (N) in phase A
#define BNT 64    // d tile (N) in phase B
#define KT 64     // K block per stage (2 wmma k-steps of 32)
#define LSTR 72   // padded LDS row stride in ushorts (144B, 16B aligned)

typedef __bf16 bf16_t;
typedef bf16_t v16bf __attribute__((ext_vector_type(16)));
typedef float v8f __attribute__((ext_vector_type(8)));
typedef unsigned short u16x8 __attribute__((ext_vector_type(8)));
typedef unsigned short u16x16 __attribute__((ext_vector_type(16)));

__device__ __forceinline__ unsigned short f2bf(float f) {
  union { float f; unsigned u; } c; c.f = f;
  unsigned u = c.u;
  u += 0x7FFFu + ((u >> 16) & 1u);   // round to nearest even
  return (unsigned short)(u >> 16);
}

// Low 32 bits of a generic pointer to __shared__ = LDS byte offset.
__device__ __forceinline__ unsigned lds_off(const void* p) {
  return (unsigned)(size_t)p;
}

// CDNA5 async copy: 16 bytes global -> LDS, tracked by ASYNCcnt.
__device__ __forceinline__ void async_b128(unsigned lds, const void* g) {
  asm volatile("global_load_async_to_lds_b128 %0, %1, off"
               :: "v"(lds), "v"(g) : "memory");
}
#define WAIT_ASYNC(n) asm volatile("s_wait_asynccnt " #n ::: "memory")

// Build a 16xbf16 WMMA fragment from two 16B-aligned LDS chunks (2x ds_load_b128).
__device__ __forceinline__ v16bf frag16(const unsigned short* p0,
                                        const unsigned short* p1) {
  u16x8 lo = *(const u16x8*)p0;
  u16x8 hi = *(const u16x8*)p1;
  u16x16 u = __builtin_shufflevector(lo, hi, 0, 1, 2, 3, 4, 5, 6, 7,
                                     8, 9, 10, 11, 12, 13, 14, 15);
  return __builtin_bit_cast(v16bf, u);
}

#define WMMA_BF16(a, b, c)                                                    \
  __builtin_amdgcn_wmma_f32_16x16x32_bf16(false, (a), false, (b), (short)0,   \
                                          (c), false, false)

// ---------------------------------------------------------------- converts --
__global__ void cvt_f32_bf16(const float* __restrict__ src,
                             unsigned short* __restrict__ dst, int n) {
  int i = blockIdx.x * blockDim.x + threadIdx.x;
  int stride = gridDim.x * blockDim.x;
  for (; i < n; i += stride) dst[i] = f2bf(src[i]);
}

// ------------------------------------------------------------------- gating --
// One wave32 per token: 16 logits via lane-parallel dots + shuffle reduce,
// softmax, group-limited routing (top2 of 4 groups), top-4 experts.
__global__ __launch_bounds__(256) void gate_kernel(
    const float* __restrict__ x, const float* __restrict__ gw,
    float* __restrict__ gate) {
  int lane = threadIdx.x & 31;
  int wave = threadIdx.x >> 5;
  int t = blockIdx.x * 8 + wave;
  const float* xr = x + (size_t)t * DIM;

  float part[NEXP];
#pragma unroll
  for (int e = 0; e < NEXP; ++e) part[e] = 0.f;
  for (int j = lane; j < DIM; j += 32) {
    float xv = xr[j];
#pragma unroll
    for (int e = 0; e < NEXP; ++e) part[e] += xv * gw[e * DIM + j];
  }
  float sc[NEXP];
#pragma unroll
  for (int e = 0; e < NEXP; ++e) {
    float v = part[e];
#pragma unroll
    for (int off = 16; off > 0; off >>= 1) v += __shfl_xor(v, off, 32);
    sc[e] = v;
  }
  float m = sc[0];
#pragma unroll
  for (int e = 1; e < NEXP; ++e) m = fmaxf(m, sc[e]);
  float s = 0.f;
#pragma unroll
  for (int e = 0; e < NEXP; ++e) { sc[e] = __expf(sc[e] - m); s += sc[e]; }
  float inv = 1.f / s;
#pragma unroll
  for (int e = 0; e < NEXP; ++e) sc[e] *= inv;

  float gs[NGRP];
#pragma unroll
  for (int g = 0; g < NGRP; ++g) {
    float gm = sc[4 * g];
#pragma unroll
    for (int j = 1; j < 4; ++j) gm = fmaxf(gm, sc[4 * g + j]);
    gs[g] = gm;
  }
  int g0 = 0; float b0 = gs[0];
#pragma unroll
  for (int g = 1; g < NGRP; ++g) if (gs[g] > b0) { b0 = gs[g]; g0 = g; }
  int g1 = -1; float b1 = -1.f;
#pragma unroll
  for (int g = 0; g < NGRP; ++g)
    if (g != g0 && gs[g] > b1) { b1 = gs[g]; g1 = g; }
  unsigned grpmask = (1u << g0) | (1u << g1);

  unsigned sel = 0;
#pragma unroll
  for (int k = 0; k < 4; ++k) {
    float best = -1.f; int bidx = 0;
#pragma unroll
    for (int e = 0; e < NEXP; ++e) {
      float v = (((grpmask >> (e >> 2)) & 1u) && !((sel >> e) & 1u)) ? sc[e] : -1.f;
      if (v > best) { best = v; bidx = e; }
    }
    sel |= 1u << bidx;
  }
  if (lane == 0) {
#pragma unroll
    for (int e = 0; e < NEXP; ++e)
      gate[(size_t)t * NEXP + e] = ((sel >> e) & 1u) ? sc[e] : 0.f;
  }
}

// ------------------------------------------------------ phase A: h = gated --
// Hw[slot][t][i] = g[t,slot] * silu(x @ W1_slot^T) * (x @ W3_slot^T), bf16.
__global__ __launch_bounds__(256) void ffn13_kernel(
    const unsigned short* __restrict__ xb,
    const unsigned short* __restrict__ w1b,
    const unsigned short* __restrict__ w3b,
    const unsigned short* __restrict__ ws1b,
    const unsigned short* __restrict__ ws3b,
    const float* __restrict__ gate,
    unsigned short* __restrict__ hw) {
  __shared__ alignas(16) unsigned short Xs[2][MT * LSTR];
  __shared__ alignas(16) unsigned short W1s[2][IT * LSTR];
  __shared__ alignas(16) unsigned short W3s[2][IT * LSTR];

  int slot = blockIdx.z;
  int m0 = blockIdx.y * MT;
  int i0 = blockIdx.x * IT;
  const unsigned short* Xp = xb + (size_t)m0 * DIM;
  const unsigned short* W1p = (slot < NEXP)
      ? (w1b + ((size_t)slot * IDIM + i0) * DIM) : (ws1b + (size_t)i0 * DIM);
  const unsigned short* W3p = (slot < NEXP)
      ? (w3b + ((size_t)slot * IDIM + i0) * DIM) : (ws3b + (size_t)i0 * DIM);

  int tid = threadIdx.x;
  int lane = tid & 31;
  int wv = tid >> 5;
  int ii = wv & 3;                      // N tile 0..3
  int mh = wv >> 2;                     // M half: covers M tiles {2mh, 2mh+1}
  int lr = lane & 15;
  int koff = (lane < 16) ? 0 : 8;       // ISA K half-split at lane 16

  // Uniform staging: every 64x64 tile = 2 async b128 per thread.
  int sr = tid >> 3;                    // rows 0..31 (+32 for 2nd chunk)
  int sq = (tid & 7) * 8;               // 8-ushort (16B) column chunk
  unsigned o0 = (unsigned)(sr * LSTR + sq) * 2;
  unsigned o1 = (unsigned)((sr + 32) * LSTR + sq) * 2;
  unsigned xbase = lds_off(&Xs[0][0]);
  unsigned w1base = lds_off(&W1s[0][0]);
  unsigned w3base = lds_off(&W3s[0][0]);
  const unsigned bstep = (unsigned)(MT * LSTR) * 2;   // per-buffer byte stride

  auto stage = [&](int buf, int k0) {
    unsigned bo = (unsigned)buf * bstep;
    async_b128(xbase + bo + o0, Xp + (size_t)sr * DIM + k0 + sq);
    async_b128(xbase + bo + o1, Xp + (size_t)(sr + 32) * DIM + k0 + sq);
    async_b128(w1base + bo + o0, W1p + (size_t)sr * DIM + k0 + sq);
    async_b128(w1base + bo + o1, W1p + (size_t)(sr + 32) * DIM + k0 + sq);
    async_b128(w3base + bo + o0, W3p + (size_t)sr * DIM + k0 + sq);
    async_b128(w3base + bo + o1, W3p + (size_t)(sr + 32) * DIM + k0 + sq);
  };

  v8f acc1[2] = {v8f{}, v8f{}};
  v8f acc3[2] = {v8f{}, v8f{}};

  const int NSTEP = DIM / KT;   // 16
  stage(0, 0);
  for (int step = 0; step < NSTEP; ++step) {
    int cur = step & 1;
    if (step + 1 < NSTEP) {
      stage(cur ^ 1, (step + 1) * KT);
      WAIT_ASYNC(6);            // in-order: current buffer complete
    } else {
      WAIT_ASYNC(0);
    }
    __syncthreads();
    const unsigned short* Xb = Xs[cur];
    const unsigned short* B1 = W1s[cur];
    const unsigned short* B3 = W3s[cur];
    int arow0 = (mh * 32 + lr) * LSTR;
    int arow1 = (mh * 32 + 16 + lr) * LSTR;
    int brow = (ii * 16 + lr) * LSTR;
#pragma unroll
    for (int s = 0; s < 2; ++s) {                 // two 16x16x32 k-steps
      int cb = s * 32 + koff;
      v16bf a0 = frag16(&Xb[arow0 + cb], &Xb[arow0 + cb + 16]);
      v16bf a1 = frag16(&Xb[arow1 + cb], &Xb[arow1 + cb + 16]);
      v16bf b1 = frag16(&B1[brow + cb], &B1[brow + cb + 16]);
      v16bf b3 = frag16(&B3[brow + cb], &B3[brow + cb + 16]);
      acc1[0] = WMMA_BF16(a0, b1, acc1[0]);
      acc1[1] = WMMA_BF16(a1, b1, acc1[1]);
      acc3[0] = WMMA_BF16(a0, b3, acc3[0]);
      acc3[1] = WMMA_BF16(a1, b3, acc3[1]);
    }
    __syncthreads();            // WAR: buffer reused at step+2
  }

  int i_n = i0 + ii * 16 + lr;
#pragma unroll
  for (int u = 0; u < 2; ++u) {
#pragma unroll
    for (int r = 0; r < 8; ++r) {
      int t = m0 + (mh * 2 + u) * 16 + r + ((lane < 16) ? 0 : 8);
      float g = (slot < NEXP) ? gate[(size_t)t * NEXP + slot] : 1.0f;
      float h1 = acc1[u][r];
      float h = (h1 / (1.f + __expf(-h1))) * acc3[u][r] * g;
      hw[((size_t)slot * TOK + t) * IDIM + i_n] = f2bf(h);
    }
  }
}

// -------------------------------------------------- phase B: y = sum H W2^T --
__global__ __launch_bounds__(256) void ffn2_kernel(
    const unsigned short* __restrict__ hw,
    const unsigned short* __restrict__ w2b,
    const unsigned short* __restrict__ ws2b,
    float* __restrict__ out) {
  __shared__ alignas(16) unsigned short Hs[2][MT * LSTR];
  __shared__ alignas(16) unsigned short W2s[2][BNT * LSTR];

  int m0 = blockIdx.y * MT;
  int d0 = blockIdx.x * BNT;
  int tid = threadIdx.x;
  int lane = tid & 31;
  int wv = tid >> 5;
  int ni = wv & 3;
  int mh = wv >> 2;
  int lr = lane & 15;
  int koff = (lane < 16) ? 0 : 8;

  int sr = tid >> 3;
  int sq = (tid & 7) * 8;
  unsigned o0 = (unsigned)(sr * LSTR + sq) * 2;
  unsigned o1 = (unsigned)((sr + 32) * LSTR + sq) * 2;
  unsigned hbase = lds_off(&Hs[0][0]);
  unsigned wbase = lds_off(&W2s[0][0]);
  const unsigned bstep = (unsigned)(MT * LSTR) * 2;

  const int KS = IDIM / KT;                 // 8 k-stages per slot
  const int NSTEP = NSLOT * KS;             // 136

  auto stage = [&](int buf, int stepv) {
    int slot_ = stepv >> 3;                 // stepv / KS
    int k0_ = (stepv & 7) * KT;
    const unsigned short* Hp = hw + ((size_t)slot_ * TOK + m0) * IDIM + k0_;
    const unsigned short* W2p = ((slot_ < NEXP)
        ? (w2b + (size_t)slot_ * DIM * IDIM) : ws2b) + (size_t)d0 * IDIM + k0_;
    unsigned bo = (unsigned)buf * bstep;
    async_b128(hbase + bo + o0, Hp + (size_t)sr * IDIM + sq);
    async_b128(hbase + bo + o1, Hp + (size_t)(sr + 32) * IDIM + sq);
    async_b128(wbase + bo + o0, W2p + (size_t)sr * IDIM + sq);
    async_b128(wbase + bo + o1, W2p + (size_t)(sr + 32) * IDIM + sq);
  };

  v8f acc[2] = {v8f{}, v8f{}};

  stage(0, 0);
  for (int step = 0; step < NSTEP; ++step) {
    int cur = step & 1;
    if (step + 1 < NSTEP) {
      stage(cur ^ 1, step + 1);
      WAIT_ASYNC(4);
    } else {
      WAIT_ASYNC(0);
    }
    __syncthreads();
    const unsigned short* Hb = Hs[cur];
    const unsigned short* Wb = W2s[cur];
    int arow0 = (mh * 32 + lr) * LSTR;
    int arow1 = (mh * 32 + 16 + lr) * LSTR;
    int brow = (ni * 16 + lr) * LSTR;
#pragma unroll
    for (int s = 0; s < 2; ++s) {
      int cb = s * 32 + koff;
      v16bf a0 = frag16(&Hb[arow0 + cb], &Hb[arow0 + cb + 16]);
      v16bf a1 = frag16(&Hb[arow1 + cb], &Hb[arow1 + cb + 16]);
      v16bf b = frag16(&Wb[brow + cb], &Wb[brow + cb + 16]);
      acc[0] = WMMA_BF16(a0, b, acc[0]);
      acc[1] = WMMA_BF16(a1, b, acc[1]);
    }
    __syncthreads();
  }

  int d = d0 + ni * 16 + lr;
#pragma unroll
  for (int u = 0; u < 2; ++u) {
#pragma unroll
    for (int r = 0; r < 8; ++r) {
      int t = m0 + (mh * 2 + u) * 16 + r + ((lane < 16) ? 0 : 8);
      out[(size_t)t * DIM + d] = acc[u][r];
    }
  }
}

// ----------------------------------------------------------------- launcher --
extern "C" void kernel_launch(void* const* d_in, const int* in_sizes, int n_in,
                              void* d_out, int out_size, void* d_ws, size_t ws_size,
                              hipStream_t stream) {
  (void)in_sizes; (void)n_in; (void)out_size; (void)ws_size;
  const float* x   = (const float*)d_in[0];
  const float* gw  = (const float*)d_in[1];
  const float* w1  = (const float*)d_in[2];
  const float* w2  = (const float*)d_in[3];
  const float* w3  = (const float*)d_in[4];
  const float* ws1 = (const float*)d_in[5];
  const float* ws2 = (const float*)d_in[6];
  const float* ws3 = (const float*)d_in[7];
  float* out = (float*)d_out;

  char* p = (char*)d_ws;
  unsigned short* xb = (unsigned short*)p;  p += (size_t)TOK * DIM * 2;
  unsigned short* w1b = (unsigned short*)p; p += (size_t)NEXP * IDIM * DIM * 2;
  unsigned short* w3b = (unsigned short*)p; p += (size_t)NEXP * IDIM * DIM * 2;
  unsigned short* w2b = (unsigned short*)p; p += (size_t)NEXP * DIM * IDIM * 2;
  unsigned short* ws1b = (unsigned short*)p; p += (size_t)IDIM * DIM * 2;
  unsigned short* ws3b = (unsigned short*)p; p += (size_t)IDIM * DIM * 2;
  unsigned short* ws2b = (unsigned short*)p; p += (size_t)DIM * IDIM * 2;
  float* gate = (float*)p;                  p += (size_t)TOK * NEXP * 4;
  unsigned short* hw = (unsigned short*)p;  p += (size_t)NSLOT * TOK * IDIM * 2;

  cvt_f32_bf16<<<1024, 256, 0, stream>>>(x,   xb,   TOK * DIM);
  cvt_f32_bf16<<<2048, 256, 0, stream>>>(w1,  w1b,  NEXP * IDIM * DIM);
  cvt_f32_bf16<<<2048, 256, 0, stream>>>(w3,  w3b,  NEXP * IDIM * DIM);
  cvt_f32_bf16<<<2048, 256, 0, stream>>>(w2,  w2b,  NEXP * DIM * IDIM);
  cvt_f32_bf16<<<512,  256, 0, stream>>>(ws1, ws1b, IDIM * DIM);
  cvt_f32_bf16<<<512,  256, 0, stream>>>(ws3, ws3b, IDIM * DIM);
  cvt_f32_bf16<<<512,  256, 0, stream>>>(ws2, ws2b, DIM * IDIM);

  gate_kernel<<<TOK / 8, 256, 0, stream>>>(x, gw, gate);

  dim3 gA(IDIM / IT, TOK / MT, NSLOT);          // (8, 32, 17)
  ffn13_kernel<<<gA, 256, 0, stream>>>(xb, w1b, w3b, ws1b, ws3b, gate, hw);

  dim3 gB(DIM / BNT, TOK / MT);                 // (16, 32)
  ffn2_kernel<<<gB, 256, 0, stream>>>(hw, w2b, ws2b, out);
}
